// MinimalAttention_4595615007285
// MI455X (gfx1250) — matrix operand order repaired and checked by reading, not compiled, not run on hardware
//
#include <hip/hip_runtime.h>

typedef __attribute__((ext_vector_type(16))) __bf16 v16bf;
typedef __attribute__((ext_vector_type(8)))  float  v8f;

#define WMMA_BF16(a, b, c) \
  __builtin_amdgcn_wmma_f32_16x16x32_bf16(false, (a), false, (b), (short)0, (c), false, false)

constexpr int kHidden = 2048;
constexpr int kHeads  = 32;
constexpr int kKV     = 8;
constexpr int kHd     = 64;
constexpr int kSeq    = 2048;
constexpr int kBatch  = 2;
constexpr int kTokens = kBatch * kSeq;   // 4096
constexpr int kKVCols = kKV * kHd;       // 512

// ---------------------------------------------------------------------------
// Fragment layouts (wave32). ln = lane & 15, half = lane >> 4.
// A-matrix 16x32 bf16 fragment (lane -> row M, ISA 16-bit A layout):
//   elements 0..7  : K = kb + 8*half + e      (pointer pre-offset by 8*half)
//   elements 8..15 : K = kb + 16 + 8*half + e (second load at +16 elements)
// B-matrix 32x16 bf16 fragment (lane -> col N, same K layout as A):
//   elements 0..7  : K = kb + 8*half + e      (pointer pre-offset by 8*half)
//   elements 8..15 : K = kb + 16 + 8*half + e (second load at +16 elements)
// Both are two aligned 16-byte loads per lane.
// ---------------------------------------------------------------------------
__device__ __forceinline__ v16bf frag_at(const __bf16* p, int second_off) {
  union { uint4 u[2]; v16bf v; } x;
  x.u[0] = *(const uint4*)(p);
  x.u[1] = *(const uint4*)(p + second_off);
  return x.v;
}

__device__ __forceinline__ v16bf load_frag_a(const __bf16* base, int ld, int row,
                                             int kb, int half) {
  return frag_at(base + (size_t)row * ld + kb + half * 8, 16);
}

__device__ __forceinline__ v16bf load_frag_b(const __bf16* base, int ld, int row,
                                             int kb, int half) {
  return frag_at(base + (size_t)row * ld + kb + half * 8, 16);
}

// ---------------------------------------------------------------------------
// Prep kernels
// ---------------------------------------------------------------------------
__global__ void f32_to_bf16_kernel(const float* __restrict__ in,
                                   __bf16* __restrict__ out, int n) {
  int i = (blockIdx.x * blockDim.x + threadIdx.x) * 4;
  if (i < n) {
    float4 f = *(const float4*)(in + i);
    out[i + 0] = (__bf16)f.x;
    out[i + 1] = (__bf16)f.y;
    out[i + 2] = (__bf16)f.z;
    out[i + 3] = (__bf16)f.w;
  }
}

// in: fp32 [K][N] row-major -> out: bf16 [N][K] row-major (W^T, K-contiguous)
__global__ void transpose_to_bf16_kernel(const float* __restrict__ in,
                                         __bf16* __restrict__ out, int K, int N) {
  int kk = blockIdx.x * 32 + threadIdx.x;
  int n  = blockIdx.y * 8  + threadIdx.y;
  if (kk < K && n < N)
    out[(size_t)n * K + kk] = (__bf16)in[(size_t)kk * N + n];
}

// ---------------------------------------------------------------------------
// GEMM: Y[row][col] = sum_k A[row][k] * BT[col][k]
// One wave computes a 64(M) x 32(N) tile; 8 WMMAs per K-step of 32.
// Fragment base pointers are hoisted and bumped by 32 elements per step so
// the loop body is pure {global_load_b128 x12, v_wmma x8, prefetch x2}.
// mode 0: q  -> bf16 [b][32][s][64]
// mode 1: k  -> bf16 [b][8][s][64]
// mode 2: v  -> bf16 [b][8][64][s]   (transposed for the PV B-operand)
// mode 3: fp32 [row][2048] (final output)
// ---------------------------------------------------------------------------
__global__ void gemm_bf16_kernel(const __bf16* __restrict__ A,
                                 const __bf16* __restrict__ BT,
                                 int Kdim, int Ncols,
                                 __bf16* __restrict__ outb,
                                 float* __restrict__ outf, int mode) {
  const int lane = threadIdx.x & 31;
  const int ln = lane & 15, half = lane >> 4;
  const int gw = (blockIdx.x * blockDim.x + threadIdx.x) >> 5;
  const int ntiles = Ncols >> 5;
  const int mt = gw / ntiles, nt = gw % ntiles;
  const int rowbase = mt * 64, colbase = nt * 32;

  const __bf16* pa0 = A + (size_t)(rowbase + ln) * Kdim + half * 8;
  const __bf16* pa1 = pa0 + (size_t)16 * Kdim;
  const __bf16* pa2 = pa0 + (size_t)32 * Kdim;
  const __bf16* pa3 = pa0 + (size_t)48 * Kdim;
  const __bf16* pb0 = BT + (size_t)(colbase + ln) * Kdim + half * 8;
  const __bf16* pb1 = pb0 + (size_t)16 * Kdim;

  const v8f vzero = {0.f, 0.f, 0.f, 0.f, 0.f, 0.f, 0.f, 0.f};
  v8f acc[4][2];
#pragma unroll
  for (int mi = 0; mi < 4; mi++)
#pragma unroll
    for (int ni = 0; ni < 2; ni++) acc[mi][ni] = vzero;

  for (int kb = 0; kb < Kdim; kb += 32) {
    // prefetch ~4 K-steps ahead off live pointers (global_prefetch_b8 w/ imm offset)
    __builtin_prefetch(pa0 + 128, 0, 3);
    __builtin_prefetch(pb0 + 128, 0, 3);

    v16bf a[4], b[2];
    a[0] = frag_at(pa0, 16);
    a[1] = frag_at(pa1, 16);
    a[2] = frag_at(pa2, 16);
    a[3] = frag_at(pa3, 16);
    b[0] = frag_at(pb0, 16);
    b[1] = frag_at(pb1, 16);
#pragma unroll
    for (int mi = 0; mi < 4; mi++)
#pragma unroll
      for (int ni = 0; ni < 2; ni++)
        acc[mi][ni] = WMMA_BF16(a[mi], b[ni], acc[mi][ni]);

    pa0 += 32; pa1 += 32; pa2 += 32; pa3 += 32;
    pb0 += 32; pb1 += 32;
  }

  // C/D layout: element r of lane -> row (r + 8*half), col ln (within 16x16 tile)
#pragma unroll
  for (int mi = 0; mi < 4; mi++)
#pragma unroll
    for (int ni = 0; ni < 2; ni++)
#pragma unroll
      for (int r = 0; r < 8; r++) {
        float val = acc[mi][ni][r];
        int row = rowbase + 16 * mi + r + 8 * half;
        int col = colbase + 16 * ni + ln;
        int bb = row >> 11, s = row & 2047;
        if (mode == 0) {
          int h = col >> 6, d = col & 63;
          outb[((((size_t)bb * kHeads + h) * kSeq + s) << 6) + d] = (__bf16)val;
        } else if (mode == 1) {
          int h = col >> 6, d = col & 63;
          outb[((((size_t)bb * kKV + h) * kSeq + s) << 6) + d] = (__bf16)val;
        } else if (mode == 2) {
          int h = col >> 6, d = col & 63;
          outb[(((size_t)bb * kKV + h) * kHd + d) * kSeq + s] = (__bf16)val;
        } else {
          outf[(size_t)row * kHidden + col] = val;
        }
      }
}

// ---------------------------------------------------------------------------
// Flash attention: one wave per (b, h, 16-query tile), 64-key chunks.
// q: [b][32][s][64], k: [b][8][s][64], vt: [b][8][64][s], ao: [b][s][2048] bf16
// ---------------------------------------------------------------------------
__global__ void attn_kernel(const __bf16* __restrict__ q,
                            const __bf16* __restrict__ kmat,
                            const __bf16* __restrict__ vt,
                            __bf16* __restrict__ ao) {
  __shared__ __align__(16) __bf16 pbuf[8][16][72];  // per-wave P tile, padded
  const int lane = threadIdx.x & 31;
  const int ln = lane & 15, half = lane >> 4;
  const int widx = threadIdx.x >> 5;
  const int gw = blockIdx.x * 8 + widx;
  const int qt = gw & 127;          // kSeq/16 = 128 query tiles
  const int h  = (gw >> 7) & 31;
  const int b  = gw >> 12;
  const int kvh = h >> 2;           // repeat_interleave(4): consecutive copies

  const __bf16* qp = q    + (((size_t)b * kHeads + h)   * kSeq << 6);
  const __bf16* kp = kmat + (((size_t)b * kKV   + kvh) * kSeq << 6);
  const __bf16* vp = vt   + (((size_t)b * kKV   + kvh) * kHd) * kSeq;

  // Q fragments with 1/sqrt(64) folded in (exact power-of-two scale in bf16)
  v16bf qf[2];
#pragma unroll
  for (int kk = 0; kk < 2; kk++) {
    qf[kk] = load_frag_a(qp, kHd, qt * 16 + ln, 32 * kk, half);
#pragma unroll
    for (int e = 0; e < 16; e++) qf[kk][e] = (__bf16)((float)qf[kk][e] * 0.125f);
  }

  const v8f vzero = {0.f, 0.f, 0.f, 0.f, 0.f, 0.f, 0.f, 0.f};
  v8f oacc[4];
#pragma unroll
  for (int t = 0; t < 4; t++) oacc[t] = vzero;
  float mrow[8], lrow[8];
#pragma unroll
  for (int r = 0; r < 8; r++) { mrow[r] = -1e30f; lrow[r] = 0.f; }

  __bf16* pb = &pbuf[widx][0][0];

  for (int kb = 0; kb < kSeq; kb += 64) {
    // prefetch next chunk of K (speculative; OOB prefetches are dropped)
    __builtin_prefetch(kp + (size_t)(kb + 64 + ln) * kHd, 0, 3);

    // --- scores: 4 key-tiles x (d = 2 K-steps of 32) ---
    v8f sc[4];
#pragma unroll
    for (int t = 0; t < 4; t++) {
      v16bf k0 = load_frag_b(kp, kHd, kb + 16 * t + ln, 0,  half);
      v16bf k1 = load_frag_b(kp, kHd, kb + 16 * t + ln, 32, half);
      v8f c = vzero;
      c = WMMA_BF16(qf[0], k0, c);
      c = WMMA_BF16(qf[1], k1, c);
      sc[t] = c;
    }
    // --- online softmax: chunk row-max, reduce across the 16 lanes of a half ---
    float nm[8];
#pragma unroll
    for (int r = 0; r < 8; r++) {
      float v = fmaxf(fmaxf(sc[0][r], sc[1][r]), fmaxf(sc[2][r], sc[3][r]));
      nm[r] = v;
    }
#pragma unroll
    for (int m = 1; m < 16; m <<= 1)
#pragma unroll
      for (int r = 0; r < 8; r++) nm[r] = fmaxf(nm[r], __shfl_xor(nm[r], m, 32));

    float sf[8];
#pragma unroll
    for (int r = 0; r < 8; r++) {
      float newm = fmaxf(nm[r], mrow[r]);
      sf[r] = __expf(mrow[r] - newm);
      mrow[r] = newm;
      lrow[r] *= sf[r];
    }
#pragma unroll
    for (int t = 0; t < 4; t++)
#pragma unroll
      for (int r = 0; r < 8; r++) oacc[t][r] *= sf[r];  // lane-local rescale

    // --- exp, row-sum, and P -> LDS (C-layout scatter: row r+8*half, col 16t+ln)
    float rs[8];
#pragma unroll
    for (int r = 0; r < 8; r++) rs[r] = 0.f;
#pragma unroll
    for (int t = 0; t < 4; t++)
#pragma unroll
      for (int r = 0; r < 8; r++) {
        float pv = __expf(sc[t][r] - mrow[r]);
        rs[r] += pv;
        pb[(r + 8 * half) * 72 + 16 * t + ln] = (__bf16)pv;
      }
#pragma unroll
    for (int m = 1; m < 16; m <<= 1)
#pragma unroll
      for (int r = 0; r < 8; r++) rs[r] += __shfl_xor(rs[r], m, 32);
#pragma unroll
    for (int r = 0; r < 8; r++) lrow[r] += rs[r];

    asm volatile("s_wait_dscnt 0" ::: "memory");

    // --- P x V: A = P (from LDS, A-layout), B = V^T rows (d, keys-contiguous)
#pragma unroll
    for (int kk = 0; kk < 2; kk++) {
      v16bf pf = load_frag_a(pb, 72, ln, 32 * kk, half);
#pragma unroll
      for (int t = 0; t < 4; t++) {
        v16bf vf = load_frag_b(vp, kSeq, 16 * t + ln, kb + 32 * kk, half);
        oacc[t] = WMMA_BF16(pf, vf, oacc[t]);
      }
    }
    asm volatile("s_wait_dscnt 0" ::: "memory");
  }

  // --- finalize and store to [b][s][h*64 + d] bf16 ---
#pragma unroll
  for (int t = 0; t < 4; t++)
#pragma unroll
    for (int r = 0; r < 8; r++) {
      float val = oacc[t][r] / lrow[r];
      int s = qt * 16 + r + 8 * half;
      int col = h * 64 + 16 * t + ln;
      ao[((size_t)b * kSeq + s) * kHidden + col] = (__bf16)val;
    }
}

// ---------------------------------------------------------------------------
extern "C" void kernel_launch(void* const* d_in, const int* in_sizes, int n_in,
                              void* d_out, int out_size, void* d_ws, size_t ws_size,
                              hipStream_t stream) {
  const float* x  = (const float*)d_in[0];
  const float* Wq = (const float*)d_in[1];
  const float* Wk = (const float*)d_in[2];
  const float* Wv = (const float*)d_in[3];
  const float* Wo = (const float*)d_in[4];
  float* out = (float*)d_out;

  char* p = (char*)d_ws;
  __bf16* xb   = (__bf16*)p; p += (size_t)kTokens * kHidden * 2;  // 16 MB
  __bf16* WqT  = (__bf16*)p; p += (size_t)kHidden * kHidden * 2;  //  8 MB
  __bf16* WkT  = (__bf16*)p; p += (size_t)kKVCols * kHidden * 2;  //  2 MB
  __bf16* WvT  = (__bf16*)p; p += (size_t)kKVCols * kHidden * 2;  //  2 MB
  __bf16* WoT  = (__bf16*)p; p += (size_t)kHidden * kHidden * 2;  //  8 MB
  __bf16* qb   = (__bf16*)p; p += (size_t)kTokens * kHidden * 2;  // 16 MB
  __bf16* kbuf = (__bf16*)p; p += (size_t)kTokens * kKVCols * 2;  //  4 MB
  __bf16* vtb  = (__bf16*)p; p += (size_t)kTokens * kKVCols * 2;  //  4 MB
  __bf16* aob  = (__bf16*)p; p += (size_t)kTokens * kHidden * 2;  // 16 MB

  // 1. x -> bf16
  {
    int n = kTokens * kHidden;
    f32_to_bf16_kernel<<<n / 4 / 256, 256, 0, stream>>>(x, xb, n);
  }
  // 2. weights -> bf16 transposed [N][K]
  dim3 tb(32, 8);
  transpose_to_bf16_kernel<<<dim3(kHidden / 32, kHidden / 8), tb, 0, stream>>>(Wq, WqT, kHidden, kHidden);
  transpose_to_bf16_kernel<<<dim3(kHidden / 32, kKVCols / 8), tb, 0, stream>>>(Wk, WkT, kHidden, kKVCols);
  transpose_to_bf16_kernel<<<dim3(kHidden / 32, kKVCols / 8), tb, 0, stream>>>(Wv, WvT, kHidden, kKVCols);
  transpose_to_bf16_kernel<<<dim3(kHidden / 32, kHidden / 8), tb, 0, stream>>>(Wo, WoT, kHidden, kHidden);

  // 3. projections (64x32 tile per wave, 8 waves per 256-thread block)
  gemm_bf16_kernel<<<(kTokens / 64) * (kHidden / 32) / 8, 256, 0, stream>>>(
      xb, WqT, kHidden, kHidden, qb, nullptr, 0);
  gemm_bf16_kernel<<<(kTokens / 64) * (kKVCols / 32) / 8, 256, 0, stream>>>(
      xb, WkT, kHidden, kKVCols, kbuf, nullptr, 1);
  gemm_bf16_kernel<<<(kTokens / 64) * (kKVCols / 32) / 8, 256, 0, stream>>>(
      xb, WvT, kHidden, kKVCols, vtb, nullptr, 2);

  // 4. flash attention: 8192 waves
  attn_kernel<<<(kBatch * kHeads * (kSeq / 16)) / 8, 256, 0, stream>>>(qb, kbuf, vtb, aob);

  // 5. output projection -> fp32 d_out
  gemm_bf16_kernel<<<(kTokens / 64) * (kHidden / 32) / 8, 256, 0, stream>>>(
      aob, WoT, kHidden, kHidden, nullptr, out, 3);
}
